// VQAttention_23021024707031
// MI455X (gfx1250) — matrix operand order repaired and checked by reading, not compiled, not run on hardware
//
#include <hip/hip_runtime.h>

// ---------------- problem constants ----------------
#define Bz 4
#define Tz 2048
#define Dz 1024
#define Hz 8
#define Kz 128
#define Vz 128
#define Sz 512
#define Cz 256
#define Rz 8
#define Wz 512          // 2*C
#define BTz (Bz*Tz)     // 8192
#define HKz (Hz*Kz)     // 1024
#define HVz (Hz*Vz)     // 1024
#define NEGB 1e30f
#define INV_TAU 0.08838834764831845f   // 1/sqrt(128)

typedef _Float16 h16;
typedef __attribute__((ext_vector_type(16))) _Float16 v16h;
typedef __attribute__((ext_vector_type(8)))  float    v8f;

#define DEV static __device__ __forceinline__

// ---------------- WMMA fragment helpers (wave32, 16x16x32 f16) ----------------
// A fragment: 16x32, row = lane&15, K(h) = h + ((lane&16)?16:0)
DEV v16h frag_a(const h16* A, int lda, int row0, int k0, int lane) {
  const h16* p = A + (size_t)(row0 + (lane & 15)) * lda + k0 + ((lane & 16) ? 16 : 0);
  v16h a;
#pragma unroll
  for (int i = 0; i < 8; ++i) a[i] = p[i];
#pragma unroll
  for (int i = 0; i < 8; ++i) a[8 + i] = p[8 + i];
  return a;
}
// B fragment from K-contiguous storage Bt[N,K] row-major (B[k][n] = Bt[n][k])
// col = lane&15, K(h) = h + ((lane&16)?16:0) -> 16 contiguous halves per lane
DEV v16h frag_bt(const h16* Bt, int ldb, int col0, int k0, int lane) {
  const h16* p = Bt + (size_t)(col0 + (lane & 15)) * ldb + k0 + ((lane & 16) ? 16 : 0);
  v16h b;
#pragma unroll
  for (int i = 0; i < 16; ++i) b[i] = p[i];
  return b;
}
DEV v8f wmma(v16h a, v16h b, v8f c) {
  return __builtin_amdgcn_wmma_f32_16x16x32_f16(false, a, false, b, (short)0, c, false, false);
}
DEV void copy32B(h16* dst, const h16* src) {   // 16 halves, two b128s
  *(uint4*)dst = *(const uint4*)src;
  *(uint4*)(dst + 8) = *(const uint4*)(src + 8);
}

// ---------------- small utility kernels ----------------
__global__ __launch_bounds__(256) void k_cvt(const float* __restrict__ s, h16* __restrict__ d, int n) {
  for (int i = blockIdx.x * 256 + threadIdx.x; i < n; i += gridDim.x * 256) d[i] = (h16)s[i];
}
// transpose-convert: src[Kd, N] (row-major) -> dst[N, Kd] (row-major, K-contiguous)
__global__ __launch_bounds__(256) void k_cvt_t(const float* __restrict__ s, h16* __restrict__ d,
                                               int N, int Kd) {
  int total = N * Kd;
  for (int i = blockIdx.x * 256 + threadIdx.x; i < total; i += gridDim.x * 256) {
    int n = i / Kd, k = i - n * Kd;
    d[i] = (h16)s[(size_t)k * N + n];
  }
}
__global__ __launch_bounds__(256) void k_cbnorm(const float* __restrict__ cb, float* __restrict__ hn) {
  int i = blockIdx.x * 256 + threadIdx.x;           // i over H*S rows
  if (i < Hz * Sz) {
    const float* p = cb + (size_t)i * Kz;
    float s = 0.f;
#pragma unroll 8
    for (int k = 0; k < Kz; ++k) s += p[k] * p[k];
    hn[i] = 0.5f * s;
  }
}
// RMSNorm: one block per token
__global__ __launch_bounds__(256) void k_rms(const float* __restrict__ x, const float* __restrict__ g,
                                             h16* __restrict__ xt) {
  __shared__ float red[256];
  int t = blockIdx.x, tid = threadIdx.x;
  const float* xr = x + (size_t)t * Dz;
  float s = 0.f;
  for (int i = tid; i < Dz; i += 256) { float v = xr[i]; s += v * v; }
  red[tid] = s; __syncthreads();
  for (int st = 128; st > 0; st >>= 1) { if (tid < st) red[tid] += red[tid + st]; __syncthreads(); }
  float inv = rsqrtf(red[0] * (1.0f / Dz) + 1e-6f);
  for (int i = tid; i < Dz; i += 256) xt[(size_t)t * Dz + i] = (h16)(xr[i] * inv * g[i]);
}

// ---------------- QKV projection: [BT,D] @ [D,H*K], LDS-staged tiles ----------------
// weights pre-transposed to [N, D]; z=0 -> q (+xl_u / +xl_v), z=1 -> k, z=2 -> v (+v_t)
__global__ __launch_bounds__(256) void k_qkv(const h16* __restrict__ xt,
                                             const h16* __restrict__ wqt, const h16* __restrict__ wkt,
                                             const h16* __restrict__ wvt,
                                             const float* __restrict__ xlu, const float* __restrict__ xlv,
                                             h16* __restrict__ qu, h16* __restrict__ qv,
                                             h16* __restrict__ kh, h16* __restrict__ vh,
                                             h16* __restrict__ vt) {
  __shared__ h16 sA[128 * 32];     // 128 M-rows x 32 K
  __shared__ h16 sB[64 * 32];      // 64 N-rows x 32 K  (Bt layout)
  int tid = threadIdx.x;
  int wave = tid >> 5, lane = tid & 31;
  int hi = (lane & 16) ? 8 : 0, lcol = lane & 15;
  int m0g = blockIdx.x * 128;
  int n0 = blockIdx.y * 64;
  const h16* Wt = (blockIdx.z == 0) ? wqt : (blockIdx.z == 1) ? wkt : wvt;
  int rowa = tid >> 1, ca = (tid & 1) * 16;          // A staging map (32B/thread)
  int rowb = tid >> 2, cb8 = (tid & 3) * 8;          // B staging map (16B/thread)
  v8f acc[4] = {};
  for (int kk = 0; kk < Dz; kk += 32) {
    const h16* srcA = xt + (size_t)(m0g + rowa) * Dz + kk + ca;
    copy32B(sA + rowa * 32 + ca, srcA);
    const h16* srcB = Wt + (size_t)(n0 + rowb) * Dz + kk + cb8;
    *(uint4*)(sB + rowb * 32 + cb8) = *(const uint4*)srcB;
    if (kk + 32 < Dz) {
      __builtin_prefetch(srcA + 32, 0, 1);           // -> global_prefetch_b8
      __builtin_prefetch(srcB + 32, 0, 1);
    }
    __syncthreads();
    v16h a = frag_a(sA, 32, wave * 16, 0, lane);
#pragma unroll
    for (int j = 0; j < 4; ++j) acc[j] = wmma(a, frag_bt(sB, 32, 16 * j, 0, lane), acc[j]);
    __syncthreads();
  }
#pragma unroll
  for (int j = 0; j < 4; ++j) {
#pragma unroll
    for (int v = 0; v < 8; ++v) {
      int row = m0g + wave * 16 + v + hi;    // (b*T + t)
      int gc = n0 + 16 * j + lcol;           // h*K + k
      int b = row >> 11, t = row & (Tz - 1);
      int h = gc >> 7, k = gc & 127;
      size_t o = (((size_t)(b * Hz + h)) * Tz + t) * Kz + k;
      float val = acc[j][v];
      if (blockIdx.z == 0) {
        qu[o] = (h16)(val + xlu[h * Kz + k]);
        qv[o] = (h16)(val + xlv[h * Kz + k]);
      } else if (blockIdx.z == 1) {
        kh[o] = (h16)val;
      } else {
        vh[o] = (h16)val;
        vt[(((size_t)(b * Hz + h)) * Vz + k) * Tz + t] = (h16)val;   // [B,H,V,T]
      }
    }
  }
}

// ---------------- VQ assignment + per-block cluster stats ----------------
// one workgroup per (b,h,r); argmax(k.c - 0.5|c|^2) over S, scatter counts/dv
__global__ __launch_bounds__(256) void k_vq(const h16* __restrict__ kh, const h16* __restrict__ cbh,
                                            const float* __restrict__ hn, const h16* __restrict__ vh,
                                            h16* __restrict__ khat, float* __restrict__ d1,
                                            float* __restrict__ dv) {
  __shared__ float sc[Cz * 32];
  __shared__ float d1s[Sz];
  int bhr = blockIdx.x, r = bhr % Rz, bh = bhr / Rz, h = bh % Hz;
  int wave = threadIdx.x >> 5, lane = threadIdx.x & 31;
  int hi = (lane & 16) ? 8 : 0, lcol = lane & 15;
  const h16* kB = kh + ((size_t)bh * Tz + r * Cz) * Kz;
  const h16* cB = cbh + (size_t)h * Sz * Kz;
  const float* hB = hn + (size_t)h * Sz;
  for (int i = threadIdx.x; i < Sz; i += 256) d1s[i] = 0.f;
  float best = -3.0e38f; int bidx = 0;
  for (int s0 = 0; s0 < Sz; s0 += 32) {
#pragma unroll
    for (int i = 0; i < 4; ++i) {
      int tt = i * 8 + wave;                 // 32 tiles: 16 M x 2 N
      int mt = tt >> 1, nt = tt & 1;
      v8f acc = {};
#pragma unroll
      for (int kk = 0; kk < Kz; kk += 32) {
        v16h a = frag_a(kB, Kz, mt * 16, kk, lane);
        acc = wmma(a, frag_bt(cB, Kz, s0 + nt * 16, kk, lane), acc);
      }
#pragma unroll
      for (int v = 0; v < 8; ++v) sc[(mt * 16 + v + hi) * 32 + nt * 16 + lcol] = acc[v];
    }
    __syncthreads();
    int tk = threadIdx.x;
#pragma unroll 8
    for (int c = 0; c < 32; ++c) {
      float val = sc[tk * 32 + c] - hB[s0 + c];
      if (val > best) { best = val; bidx = s0 + c; }
    }
    __syncthreads();
  }
  // quantized key = codebook row
  int tk = threadIdx.x;
  const h16* crow = cB + (size_t)bidx * Kz;
  h16* krow = khat + ((size_t)bh * Tz + r * Cz + tk) * Kz;
#pragma unroll 8
  for (int k = 0; k < Kz; ++k) krow[k] = crow[k];
  // cluster stats
  atomicAdd(&d1s[bidx], 1.0f);
  const h16* vrow = vh + ((size_t)bh * Tz + r * Cz + tk) * Vz;
  float* dvrow = dv + (((size_t)bhr) * Sz + bidx) * Vz;
#pragma unroll 8
  for (int vv = 0; vv < Vz; ++vv) atomicAdd(&dvrow[vv], (float)vrow[vv]);
  __syncthreads();
  for (int i = threadIdx.x; i < Sz; i += 256) d1[(size_t)bhr * Sz + i] = d1s[i];
}

// ---------------- cumulative cluster means + log-lower bias ----------------
// block per (b,h,s), thread per v-dim; udl stored transposed [B,H,R,V,S]
__global__ __launch_bounds__(128) void k_stats(const float* __restrict__ d1, const float* __restrict__ dv,
                                               float* __restrict__ logl, h16* __restrict__ udlt) {
  int bhs = blockIdx.x, s = bhs % Sz, bh = bhs / Sz, vv = threadIdx.x;
  float cch[Rz], cvh[Rz];
  float cc = 0.f, cv = 0.f;
#pragma unroll
  for (int r = 0; r < Rz; ++r) {
    size_t o = ((size_t)bh * Rz + r) * Sz + s;
    cc += d1[o];
    cv += dv[o * Vz + vv];
    cch[r] = cc; cvh[r] = cv;
  }
#pragma unroll
  for (int r = 0; r < Rz; ++r) {
    size_t bhr = (size_t)bh * Rz + r;
    float u = 0.f;
    if (r >= 2) u = cvh[r - 2] / fmaxf(cch[r - 2], 1.0f);
    udlt[(bhr * Vz + vv) * Sz + s] = (h16)u;
    if (vv == 0) {
      float lower = (r >= 2) ? cch[r - 2] : 0.f;
      logl[bhr * Sz + s] = (lower > 0.f) ? logf(fmaxf(lower, 1.0f)) : -NEGB;
    }
  }
}

// ---------------- positional term, pre-shifted: bds[c,j] = (q_v . xl_r[j+C-1-c]) ----------------
__global__ __launch_bounds__(256) void k_bd(const h16* __restrict__ qvh, const h16* __restrict__ xlr,
                                            h16* __restrict__ bds) {
  int bhr = blockIdx.x, r = bhr % Rz, bh = bhr / Rz, h = bh % Hz;
  int wave = threadIdx.x >> 5, lane = threadIdx.x & 31;
  int hi = (lane & 16) ? 8 : 0, lcol = lane & 15;
  const h16* qB = qvh + ((size_t)bh * Tz + r * Cz) * Kz;
  const h16* xB = xlr + (size_t)h * Wz * Kz;
  for (int i = 0; i < 64; ++i) {                 // 512 tiles / 8 waves
    int tt = wave * 64 + i;
    int mt = tt >> 5, nt = tt & 31;
    v8f acc = {};
#pragma unroll
    for (int kk = 0; kk < Kz; kk += 32) {
      v16h a = frag_a(qB, Kz, mt * 16, kk, lane);
      acc = wmma(a, frag_bt(xB, Kz, nt * 16, kk, lane), acc);
    }
#pragma unroll
    for (int v = 0; v < 8; ++v) {
      int c = mt * 16 + v + hi;
      int wr = nt * 16 + lcol;
      int j = wr - (Cz - 1 - c);                 // rel_shift
      if (j >= 0 && j < Wz)
        bds[((size_t)bh * Tz + r * Cz + c) * Wz + j] = (h16)acc[v];
    }
  }
}

// ---------------- attention with flash-style online softmax ----------------
__global__ __launch_bounds__(256) void k_attn(const h16* __restrict__ quh, const h16* __restrict__ cbh,
                                              const h16* __restrict__ khat, const h16* __restrict__ vt,
                                              const h16* __restrict__ udlt, const float* __restrict__ logl,
                                              const h16* __restrict__ bds, h16* __restrict__ attout) {
  __shared__ h16 pbuf[8][16 * 32];
  int bhr = blockIdx.x, r = bhr % Rz, bh = bhr / Rz, h = bh % Hz, b = bh / Hz;
  int wave = threadIdx.x >> 5, lane = threadIdx.x & 31;
  int hi = (lane & 16) ? 8 : 0, lcol = lane & 15;
  const h16* quB = quh + ((size_t)bh * Tz + r * Cz) * Kz;
  const h16* cB = cbh + (size_t)h * Sz * Kz;
  const h16* kB = khat + (size_t)bh * Tz * Kz;
  const h16* vtB = vt + (size_t)bh * Vz * Tz;        // [V, T] per (b,h)
  const h16* uBt = udlt + (size_t)bhr * Vz * Sz;     // [V, S] per (b,h,r)
  const float* lB = logl + (size_t)bhr * Sz;
  const h16* bdB = bds + ((size_t)bh * Tz + r * Cz) * Wz;
  h16* pb = &pbuf[wave][0];

  for (int mt = 0; mt < 2; ++mt) {
    int m0 = wave * 32 + mt * 16;
    float mrow[8], lrow[8];
    v8f acc[8] = {};
#pragma unroll
    for (int v = 0; v < 8; ++v) { mrow[v] = -3.0e38f; lrow[v] = 0.f; }

    for (int cc = 0; cc < 32; ++cc) {            // 1024 logits in chunks of 32
      int col0 = cc * 32;
      bool is_cache = (col0 < Sz);
      int j0 = col0 - Sz;
      if (is_cache) { if (r < 2) continue; }
      else {
        if (r == 0 && j0 + 32 <= Cz) continue;   // padded-prev fully masked
        if (j0 > m0 + 15 + Cz) continue;         // fully acausal for this tile
      }
      int t0 = (r - 1) * Cz + j0;                // >= 0 for every kept recent chunk
      v8f s0 = {}, s1 = {};
      if (is_cache) {
#pragma unroll
        for (int kk = 0; kk < Kz; kk += 32) {
          v16h a = frag_a(quB, Kz, m0, kk, lane);
          s0 = wmma(a, frag_bt(cB, Kz, col0, kk, lane), s0);
          s1 = wmma(a, frag_bt(cB, Kz, col0 + 16, kk, lane), s1);
        }
        float b0 = lB[col0 + lcol], b1 = lB[col0 + 16 + lcol];
#pragma unroll
        for (int v = 0; v < 8; ++v) {
          s0[v] = s0[v] * INV_TAU + b0;
          s1[v] = s1[v] * INV_TAU + b1;
        }
      } else {
#pragma unroll
        for (int kk = 0; kk < Kz; kk += 32) {
          v16h a = frag_a(quB, Kz, m0, kk, lane);
          s0 = wmma(a, frag_bt(kB, Kz, t0, kk, lane), s0);
          s1 = wmma(a, frag_bt(kB, Kz, t0 + 16, kk, lane), s1);
        }
#pragma unroll
        for (int v = 0; v < 8; ++v) {
          int i = m0 + v + hi;
          int j = j0 + lcol;
          bool ok = (j <= i + Cz) && !(r == 0 && j < Cz);
          s0[v] = ok ? (s0[v] + (float)bdB[(size_t)i * Wz + j]) * INV_TAU : -NEGB;
          j += 16;
          ok = (j <= i + Cz) && !(r == 0 && j < Cz);
          s1[v] = ok ? (s1[v] + (float)bdB[(size_t)i * Wz + j]) * INV_TAU : -NEGB;
        }
      }
      // per-row max across the 16-lane group owning each row
      float cm[8];
#pragma unroll
      for (int v = 0; v < 8; ++v) cm[v] = fmaxf(s0[v], s1[v]);
#pragma unroll
      for (int off = 1; off < 16; off <<= 1)
#pragma unroll
        for (int v = 0; v < 8; ++v) cm[v] = fmaxf(cm[v], __shfl_xor(cm[v], off, 32));
      float scale[8], ps[8];
#pragma unroll
      for (int v = 0; v < 8; ++v) {
        float mn = fmaxf(mrow[v], cm[v]);
        scale[v] = __expf(mrow[v] - mn);
        mrow[v] = mn;
        s0[v] = __expf(s0[v] - mn);
        s1[v] = __expf(s1[v] - mn);
        ps[v] = s0[v] + s1[v];
      }
#pragma unroll
      for (int off = 1; off < 16; off <<= 1)
#pragma unroll
        for (int v = 0; v < 8; ++v) ps[v] += __shfl_xor(ps[v], off, 32);
#pragma unroll
      for (int v = 0; v < 8; ++v) lrow[v] = lrow[v] * scale[v] + ps[v];
#pragma unroll
      for (int nt = 0; nt < 8; ++nt)
#pragma unroll
        for (int v = 0; v < 8; ++v) acc[nt][v] *= scale[v];
      // D-layout -> A-layout via per-wave LDS tile
#pragma unroll
      for (int v = 0; v < 8; ++v) {
        int rr = v + hi;
        pb[rr * 32 + lcol] = (h16)s0[v];
        pb[rr * 32 + 16 + lcol] = (h16)s1[v];
      }
      __builtin_amdgcn_wave_barrier();
      asm volatile("s_wait_dscnt 0" ::: "memory");   // same-wave LDS store->load ordering
      v16h pa = frag_a(pb, 32, 0, 0, lane);
      if (is_cache) {
#pragma unroll
        for (int nt = 0; nt < 8; ++nt)
          acc[nt] = wmma(pa, frag_bt(uBt, Sz, nt * 16, col0, lane), acc[nt]);
      } else {
#pragma unroll
        for (int nt = 0; nt < 8; ++nt)
          acc[nt] = wmma(pa, frag_bt(vtB, Tz, nt * 16, t0, lane), acc[nt]);
      }
      __builtin_amdgcn_wave_barrier();
    }
    // normalize and store to [B, T, H*V] (A-operand layout for the output GEMM)
#pragma unroll
    for (int v = 0; v < 8; ++v) {
      float invl = 1.0f / lrow[v];
      int t = r * Cz + m0 + v + hi;
#pragma unroll
      for (int nt = 0; nt < 8; ++nt)
        attout[((size_t)b * Tz + t) * HVz + h * Vz + nt * 16 + lcol] = (h16)(acc[nt][v] * invl);
    }
  }
}

// ---------------- output projection: [BT, H*V] @ [H*V, D] -> f32, LDS-staged ----------------
__global__ __launch_bounds__(256) void k_oproj(const h16* __restrict__ attout, const h16* __restrict__ wot,
                                               float* __restrict__ out) {
  __shared__ h16 sA[128 * 32];
  __shared__ h16 sB[64 * 32];
  int tid = threadIdx.x;
  int wave = tid >> 5, lane = tid & 31;
  int hi = (lane & 16) ? 8 : 0, lcol = lane & 15;
  int m0g = blockIdx.x * 128;
  int n0 = blockIdx.y * 64;
  int rowa = tid >> 1, ca = (tid & 1) * 16;
  int rowb = tid >> 2, cb8 = (tid & 3) * 8;
  v8f acc[4] = {};
  for (int kk = 0; kk < HVz; kk += 32) {
    const h16* srcA = attout + (size_t)(m0g + rowa) * HVz + kk + ca;
    copy32B(sA + rowa * 32 + ca, srcA);
    const h16* srcB = wot + (size_t)(n0 + rowb) * HVz + kk + cb8;
    *(uint4*)(sB + rowb * 32 + cb8) = *(const uint4*)srcB;
    if (kk + 32 < HVz) {
      __builtin_prefetch(srcA + 32, 0, 1);
      __builtin_prefetch(srcB + 32, 0, 1);
    }
    __syncthreads();
    v16h a = frag_a(sA, 32, wave * 16, 0, lane);
#pragma unroll
    for (int j = 0; j < 4; ++j) acc[j] = wmma(a, frag_bt(sB, 32, 16 * j, 0, lane), acc[j]);
    __syncthreads();
  }
#pragma unroll
  for (int j = 0; j < 4; ++j)
#pragma unroll
    for (int v = 0; v < 8; ++v)
      out[(size_t)(m0g + wave * 16 + v + hi) * Dz + n0 + 16 * j + lcol] = acc[j][v];
}

// ---------------- launcher ----------------
extern "C" void kernel_launch(void* const* d_in, const int* in_sizes, int n_in,
                              void* d_out, int out_size, void* d_ws, size_t ws_size,
                              hipStream_t stream) {
  const float* x    = (const float*)d_in[0];
  const float* g_ln = (const float*)d_in[1];
  const float* Wq   = (const float*)d_in[2];
  const float* Wk   = (const float*)d_in[3];
  const float* Wv   = (const float*)d_in[4];
  const float* cb   = (const float*)d_in[5];
  const float* xlr  = (const float*)d_in[6];
  const float* xlu  = (const float*)d_in[7];
  const float* xlv  = (const float*)d_in[8];
  const float* Wo   = (const float*)d_in[9];
  float* out = (float*)d_out;

  char* ws = (char*)d_ws;
  size_t off = 0;
  auto alloc = [&](size_t bytes) { char* p = ws + off; off = (off + bytes + 255) & ~(size_t)255; return p; };
  h16*  xt    = (h16*)alloc((size_t)BTz * Dz * 2);
  h16*  wq_t  = (h16*)alloc((size_t)HKz * Dz * 2);   // [H*K, D]
  h16*  wk_t  = (h16*)alloc((size_t)HKz * Dz * 2);
  h16*  wv_t  = (h16*)alloc((size_t)HVz * Dz * 2);
  h16*  wo_t  = (h16*)alloc((size_t)Dz * HVz * 2);   // [D, H*V]
  h16*  cb_h  = (h16*)alloc((size_t)Hz * Sz * Kz * 2);
  h16*  xlr_h = (h16*)alloc((size_t)Hz * Wz * Kz * 2);
  float* hn   = (float*)alloc((size_t)Hz * Sz * 4);
  h16*  qu_h  = (h16*)alloc((size_t)Bz * Hz * Tz * Kz * 2);
  h16*  qv_h  = (h16*)alloc((size_t)Bz * Hz * Tz * Kz * 2);
  h16*  k_h   = (h16*)alloc((size_t)Bz * Hz * Tz * Kz * 2);
  h16*  v_h   = (h16*)alloc((size_t)Bz * Hz * Tz * Vz * 2);
  h16*  v_t   = (h16*)alloc((size_t)Bz * Hz * Vz * Tz * 2);  // [B,H,V,T]
  h16*  khat  = (h16*)alloc((size_t)Bz * Hz * Tz * Kz * 2);
  float* d1   = (float*)alloc((size_t)Bz * Hz * Rz * Sz * 4);
  float* dv   = (float*)alloc((size_t)Bz * Hz * Rz * Sz * Vz * 4);
  float* logl = (float*)alloc((size_t)Bz * Hz * Rz * Sz * 4);
  h16*  udlt  = (h16*)alloc((size_t)Bz * Hz * Rz * Vz * Sz * 2); // [B,H,R,V,S]
  h16*  bds   = (h16*)alloc((size_t)Bz * Hz * Tz * Wz * 2);
  h16*  atto  = (h16*)alloc((size_t)Bz * Tz * HVz * 2);
  (void)ws_size; (void)in_sizes; (void)n_in; (void)out_size;

  hipMemsetAsync(dv, 0, (size_t)Bz * Hz * Rz * Sz * Vz * 4, stream);

  // weight conversions (transposed to K-contiguous where needed)
  k_cvt_t<<<4096, 256, 0, stream>>>(Wq, wq_t, HKz, Dz);
  k_cvt_t<<<4096, 256, 0, stream>>>(Wk, wk_t, HKz, Dz);
  k_cvt_t<<<4096, 256, 0, stream>>>(Wv, wv_t, HVz, Dz);
  k_cvt_t<<<4096, 256, 0, stream>>>(Wo, wo_t, Dz, HVz);
  {
    int n = Hz * Sz * Kz;
    k_cvt<<<(n + 255) / 256, 256, 0, stream>>>(cb, cb_h, n);
    n = Hz * Wz * Kz;
    k_cvt<<<(n + 255) / 256, 256, 0, stream>>>(xlr, xlr_h, n);
  }
  k_cbnorm<<<(Hz * Sz + 255) / 256, 256, 0, stream>>>(cb, hn);

  k_rms<<<BTz, 256, 0, stream>>>(x, g_ln, xt);
  k_qkv<<<dim3(BTz / 128, HKz / 64, 3), 256, 0, stream>>>(xt, wq_t, wk_t, wv_t, xlu, xlv,
                                                          qu_h, qv_h, k_h, v_h, v_t);
  k_vq<<<Bz * Hz * Rz, 256, 0, stream>>>(k_h, cb_h, hn, v_h, khat, d1, dv);
  k_stats<<<Bz * Hz * Sz, 128, 0, stream>>>(d1, dv, logl, udlt);
  k_bd<<<Bz * Hz * Rz, 256, 0, stream>>>(qv_h, xlr_h, bds);
  k_attn<<<Bz * Hz * Rz, 256, 0, stream>>>(qu_h, cb_h, khat, v_t, udlt, logl, bds, atto);
  k_oproj<<<dim3(BTz / 128, Dz / 64), 256, 0, stream>>>(atto, wo_t, out);
}
